// MultiHeadAttention_14439680049784
// MI455X (gfx1250) — compile-verified, matrix operand order checked by reading, not run on hardware
//
#include <hip/hip_runtime.h>
#include <hip/hip_bf16.h>

// MHA forward for MI455X (gfx1250), wave32 + v_wmma_f32_16x16x32_f16.
// ws usage (f16): Q [b][h][t][dh], K [b][h][t][dh], V^T [b][h][dh][t],
// ctx [b][t][h*64+dh]  -> 4 * B*H*T*DH * 2 = 32 MB.

constexpr int B_ = 4, T_ = 2048, D_ = 512, H_ = 8, DH_ = 64;

typedef __attribute__((ext_vector_type(16))) _Float16 v16h;
typedef __attribute__((ext_vector_type(8)))  _Float16 v8h;
typedef __attribute__((ext_vector_type(8)))  float    v8f;

// Async global->LDS staging path (CDNA5): guarded so absence of the builtins
// falls back to the direct-global fragment loads.
#if __has_builtin(__builtin_amdgcn_global_load_async_to_lds_b128) && \
    __has_builtin(__builtin_amdgcn_s_wait_asynccnt)
#define ATTN_ASYNC 1
#else
#define ATTN_ASYNC 0
#endif

#if ATTN_ASYNC
// Builtin prototype (from hipcc diagnostic): param0 = AS1 pointer to
// 'int __attribute__((vector_size(16)))', param1 = AS3 pointer to same,
// then imm offset and imm cpol.
typedef int v4i_ __attribute__((vector_size(16)));
typedef __attribute__((address_space(1))) v4i_* as1_v4i_p;
typedef __attribute__((address_space(3))) v4i_* as3_v4i_p;
// generic->AS casts via integer (low 32 bits of a generic LDS pointer are the
// LDS byte offset; global generic == AS1 numerically).
__device__ __forceinline__ as1_v4i_p to_as1(const void* p) {
    return (as1_v4i_p)(unsigned long long)p;
}
__device__ __forceinline__ as3_v4i_p to_as3(const void* p) {
    return (as3_v4i_p)(unsigned int)(unsigned long long)p;
}
#endif

// WMMA 16x16x32 f16 A/B operand fragment from a row-major f16 tile.
// Per-lane layout (wave32): row = lane&15; elements 0..7 hold
// k = k0 + (lane>>4)*8 + i, elements 8..15 hold k = k0 + 16 + (lane>>4)*8 + i.
// ld (halves) must be a multiple of 8, k0 a multiple of 32.
__device__ __forceinline__ v16h load_frag_f16(const _Float16* base, int ld, int k0, int lane) {
    const _Float16* p = base + (size_t)(lane & 15) * ld + k0 + ((lane >> 4) << 3);
    v8h lo = *(const v8h*)(p);
    v8h hi = *(const v8h*)(p + 16);
    v16h f;
#pragma unroll
    for (int i = 0; i < 8; ++i) { f[i] = lo[i]; f[i + 8] = hi[i]; }
    return f;
}

// Same fragment, sourced from row-major f32 and converted to f16.
__device__ __forceinline__ v16h load_frag_f32(const float* base, int ld, int k0, int lane) {
    const float* p = base + (size_t)(lane & 15) * ld + k0 + ((lane >> 4) << 3);
    v16h f;
#pragma unroll
    for (int i = 0; i < 8; ++i) { f[i] = (_Float16)p[i]; f[i + 8] = (_Float16)p[16 + i]; }
    return f;
}

__device__ __forceinline__ v8f wmma_f16(v16h a, v16h b, v8f c) {
    return __builtin_amdgcn_wmma_f32_16x16x32_f16(false, a, false, b, (short)0, c, false, false);
}

// ---------------------------------------------------------------------------
// Kernel 1: fused Q/K/V projection. grid = (D/64, B*T/128, 3), block = 256.
// Q,K stored [b][head][t][dh]; V stored transposed [b][head][dh][t] so the
// attention kernel's B fragments for P*V need no transpose.
// ---------------------------------------------------------------------------
__global__ __launch_bounds__(256) void qkv_proj_kernel(
    const float* __restrict__ q, const float* __restrict__ k, const float* __restrict__ v,
    const float* __restrict__ Wq, const float* __restrict__ bq,
    const float* __restrict__ Wk, const float* __restrict__ bk,
    const float* __restrict__ Wv, const float* __restrict__ bv,
    _Float16* __restrict__ ws)
{
    constexpr int LDW = 264;                       // 256 + 8 pad, 16B-aligned rows
    __shared__ __align__(16) _Float16 Wt[64][LDW]; // W^T tile: [n][k], f16

    const int head = blockIdx.x;   // N-block of 64 == one head
    const int mblk = blockIdx.y;   // 128 rows each
    const int sel  = blockIdx.z;   // 0=Q 1=K 2=V

    const float* X    = sel == 0 ? q  : sel == 1 ? k  : v;
    const float* W    = sel == 0 ? Wq : sel == 1 ? Wk : Wv;
    const float* bias = sel == 0 ? bq : sel == 1 ? bk : bv;
    _Float16* dst = ws + (size_t)sel * ((size_t)B_ * H_ * T_ * DH_);

    const int tid = threadIdx.x;
    const int wave = tid >> 5, lane = tid & 31;
    const int r15 = lane & 15, lh = lane >> 4;
    const int mrow = mblk * 128 + wave * 16;
    const float* Abase = X + (size_t)mrow * D_;

    v8f acc[4] = {};

    for (int kh = 0; kh < D_; kh += 256) {
        __syncthreads();
        {   // stage Wt[n][kk] = W[kh+kk][head*64+n], coalesced over n
            const int n = tid & 63, kb = tid >> 6;
#pragma unroll
            for (int i = 0; i < 64; ++i) {
                const int kk = kb + i * 4;
                Wt[n][kk] = (_Float16)W[(size_t)(kh + kk) * D_ + head * 64 + n];
            }
        }
        __syncthreads();
        if (kh + 256 < D_) {    // prefetch next weight k-block (global_prefetch_b8)
#pragma unroll
            for (int i = 0; i < 4; ++i)
                __builtin_prefetch(&W[(size_t)(kh + 256 + (tid >> 6) + i * 4) * D_ +
                                      head * 64 + (tid & 63)], 0, 1);
        }
#pragma unroll 4
        for (int ks = 0; ks < 256; ks += 32) {
            const v16h af = load_frag_f32(Abase, D_, kh + ks, lane);
#pragma unroll
            for (int j = 0; j < 4; ++j)
                acc[j] = wmma_f16(af, load_frag_f16(&Wt[j * 16][0], LDW, ks, lane), acc[j]);
        }
    }

    // bias + scatter (C layout: M = r + 8*(lane>>4), N = lane&15)
#pragma unroll
    for (int j = 0; j < 4; ++j) {
        const int dh = j * 16 + r15;
        const float bb = bias[head * 64 + dh];
#pragma unroll
        for (int r = 0; r < 8; ++r) {
            const int m = mrow + r + 8 * lh;
            const int bidx = m / T_, t = m % T_;
            const _Float16 val = (_Float16)(acc[j][r] + bb);
            if (sel == 2)  // V transposed: [b][head][dh][t]   (uniform branch)
                dst[(((size_t)bidx * H_ + head) * DH_ + dh) * T_ + t] = val;
            else           // Q,K:          [b][head][t][dh]
                dst[(((size_t)bidx * H_ + head) * T_ + t) * DH_ + dh] = val;
        }
    }
}

// ---------------------------------------------------------------------------
// Kernel 2: flash attention. grid = (T/128, B*H), block = 256.
// Each wave owns 16 query rows. If async-to-LDS builtins are available, the
// 8KB K and V^T key tiles are staged once per workgroup with
// global_load_async_to_lds_b128, double-buffered (s_wait_asynccnt + barrier);
// otherwise every wave loads its fragments directly from global (L2-resident).
// ---------------------------------------------------------------------------
#if ATTN_ASYNC
__device__ __forceinline__ void stage_kv_tile(
    const _Float16* __restrict__ Kp, const _Float16* __restrict__ Vtp, int kt,
    _Float16* Kdst, _Float16* Vdst, int tid)
{
    // tile = 64 rows x 64 halves (8KB); 512 chunks of 16B; 256 threads x 2
#pragma unroll
    for (int i = 0; i < 2; ++i) {
        const int c = tid + i * 256;
        const int row = c >> 3;
        const int col = (c & 7) << 3;
        __builtin_amdgcn_global_load_async_to_lds_b128(
            to_as1(Kp + (size_t)(kt + row) * DH_ + col),
            to_as3(Kdst + row * 64 + col), 0, 0);
        __builtin_amdgcn_global_load_async_to_lds_b128(
            to_as1(Vtp + (size_t)row * T_ + kt + col),
            to_as3(Vdst + row * 64 + col), 0, 0);
    }
}
#endif

__global__ __launch_bounds__(256) void attn_kernel(
    const _Float16* __restrict__ ws, const unsigned char* __restrict__ mask,
    _Float16* __restrict__ ctx)
{
    constexpr int LDP = 72;                           // 64 + 8 pad, 16B-aligned rows
#if ATTN_ASYNC
    __shared__ __align__(16) _Float16 Ksh[2][64][64]; // [key][dh]
    __shared__ __align__(16) _Float16 Vsh[2][64][64]; // [dh][key]
#endif
    __shared__ __align__(16) _Float16 Pl[8][16][LDP]; // per-wave P transpose buffer

    const size_t S1 = (size_t)B_ * H_ * T_ * DH_;
    const int mtile = blockIdx.x;
    const int bh = blockIdx.y;
    const int b = bh / H_, h = bh % H_;
    const _Float16* Qp  = ws + (size_t)bh * T_ * DH_;
    const _Float16* Kp  = ws + S1 + (size_t)bh * T_ * DH_;
    const _Float16* Vtp = ws + 2 * S1 + (size_t)bh * DH_ * T_;  // [dh][t]
    const unsigned char* mrow = mask + (size_t)b * T_;

    const int tid = threadIdx.x, wave = tid >> 5, lane = tid & 31;
    const int r15 = lane & 15, lh = lane >> 4;
    const int q0 = mtile * 128 + wave * 16;
    const float NEGINF = -__builtin_inff();

    // Q fragments held in registers for the whole key loop.
    const v16h qf0 = load_frag_f16(Qp + (size_t)q0 * DH_, DH_, 0, lane);
    const v16h qf1 = load_frag_f16(Qp + (size_t)q0 * DH_, DH_, 32, lane);

    v8f o[4] = {};
    float mi[8], li[8];
#pragma unroll
    for (int r = 0; r < 8; ++r) { mi[r] = NEGINF; li[r] = 0.f; }

#if ATTN_ASYNC
    stage_kv_tile(Kp, Vtp, 0, &Ksh[0][0][0], &Vsh[0][0][0], tid);   // preamble
#endif

    for (int kt = 0; kt < T_; kt += 64) {
#if ATTN_ASYNC
        const int cur = (kt >> 6) & 1;
        __builtin_amdgcn_s_wait_asynccnt(0);   // my part of buf[cur] landed
        __syncthreads();                       // everyone's part landed; prev reads done
        if (kt + 64 < T_)                      // stage next tile into the other buffer
            stage_kv_tile(Kp, Vtp, kt + 64, &Ksh[cur ^ 1][0][0], &Vsh[cur ^ 1][0][0], tid);
        const _Float16* Kt  = &Ksh[cur][0][0];
        const _Float16* Vt2 = &Vsh[cur][0][0];
        const int ldv = 64;
#else
        const int ktn = kt + 64;               // prefetch next tile (global_prefetch_b8)
        if (ktn < T_) {
            __builtin_prefetch(Kp + (size_t)(ktn + (lane << 1)) * DH_, 0, 1);
            __builtin_prefetch(Vtp + (size_t)lane * T_ + ktn, 0, 1);
            __builtin_prefetch(Vtp + (size_t)(lane + 32) * T_ + ktn, 0, 1);
        }
        const _Float16* Kt  = Kp + (size_t)kt * DH_;
        const _Float16* Vt2 = Vtp + kt;
        const int ldv = T_;
#endif

        // S = Q K^T : B fragment row = key = lane&15, k = dh (ld = 64 both paths)
        v8f s[4] = {};
#pragma unroll
        for (int j = 0; j < 4; ++j) {
            const _Float16* Kb = Kt + (size_t)(j * 16) * 64;
            s[j] = wmma_f16(qf0, load_frag_f16(Kb, 64, 0, lane), s[j]);
            s[j] = wmma_f16(qf1, load_frag_f16(Kb, 64, 32, lane), s[j]);
        }

        // scale 1/sqrt(DH) + key-padding mask (key col = kt + j*16 + lane&15)
#pragma unroll
        for (int j = 0; j < 4; ++j) {
            const bool ok = mrow[kt + j * 16 + r15] != 0;
#pragma unroll
            for (int r = 0; r < 8; ++r)
                s[j][r] = ok ? s[j][r] * 0.125f : NEGINF;
        }

        // online softmax: rows live across each 16-lane half; shuffle reductions
#pragma unroll
        for (int r = 0; r < 8; ++r) {
            float mx = fmaxf(fmaxf(s[0][r], s[1][r]), fmaxf(s[2][r], s[3][r]));
#pragma unroll
            for (int d = 1; d < 16; d <<= 1) mx = fmaxf(mx, __shfl_xor(mx, d, 16));
            const float mnew = fmaxf(mi[r], mx);
            const float alpha = (mi[r] == NEGINF) ? 0.f : __expf(mi[r] - mnew);
            float rs = 0.f;
#pragma unroll
            for (int j = 0; j < 4; ++j) {
                const float p = (s[j][r] == NEGINF) ? 0.f : __expf(s[j][r] - mnew);
                s[j][r] = p;
                rs += p;
            }
#pragma unroll
            for (int d = 1; d < 16; d <<= 1) rs += __shfl_xor(rs, d, 16);
            li[r] = li[r] * alpha + rs;
            mi[r] = mnew;
#pragma unroll
            for (int j = 0; j < 4; ++j) o[j][r] *= alpha;
        }

        // P: C-layout -> per-wave LDS -> A-layout fragments (same-wave, in-order)
#pragma unroll
        for (int j = 0; j < 4; ++j)
#pragma unroll
            for (int r = 0; r < 8; ++r)
                Pl[wave][r + 8 * lh][j * 16 + r15] = (_Float16)s[j][r];

        const v16h pf0 = load_frag_f16(&Pl[wave][0][0], LDP, 0, lane);
        const v16h pf1 = load_frag_f16(&Pl[wave][0][0], LDP, 32, lane);

        // O += P V : B fragment row = dh, k = key
#pragma unroll
        for (int j = 0; j < 4; ++j) {
            const _Float16* Vb = Vt2 + (size_t)(j * 16) * ldv;
            o[j] = wmma_f16(pf0, load_frag_f16(Vb, ldv, 0, lane), o[j]);
            o[j] = wmma_f16(pf1, load_frag_f16(Vb, ldv, 32, lane), o[j]);
        }
    }

    // normalize (all-masked rows -> 0, matches nan_to_num); ctx [b][t][h*64+dh]
#pragma unroll
    for (int j = 0; j < 4; ++j) {
#pragma unroll
        for (int r = 0; r < 8; ++r) {
            const int t = q0 + r + 8 * lh;
            const float denom = li[r];
            const float val = denom > 0.f ? o[j][r] / denom : 0.f;
            ctx[((size_t)b * T_ + t) * D_ + h * DH_ + j * 16 + r15] = (_Float16)val;
        }
    }
}

// ---------------------------------------------------------------------------
// Kernel 3: output projection. grid = (D/64, B*T/128), block = 256.
// A fragments read directly from f16 ctx; f32 output + bias.
// ---------------------------------------------------------------------------
__global__ __launch_bounds__(256) void out_proj_kernel(
    const _Float16* __restrict__ ctx, const float* __restrict__ Wo,
    const float* __restrict__ bo, float* __restrict__ out)
{
    constexpr int LDW = 264;
    __shared__ __align__(16) _Float16 Wt[64][LDW];

    const int nblk = blockIdx.x, mblk = blockIdx.y;
    const int tid = threadIdx.x, wave = tid >> 5, lane = tid & 31;
    const int r15 = lane & 15, lh = lane >> 4;
    const int mrow = mblk * 128 + wave * 16;
    const _Float16* Abase = ctx + (size_t)mrow * D_;

    v8f acc[4] = {};
    for (int kh = 0; kh < D_; kh += 256) {
        __syncthreads();
        {
            const int n = tid & 63, kb = tid >> 6;
#pragma unroll
            for (int i = 0; i < 64; ++i) {
                const int kk = kb + i * 4;
                Wt[n][kk] = (_Float16)Wo[(size_t)(kh + kk) * D_ + nblk * 64 + n];
            }
        }
        __syncthreads();
        if (kh + 256 < D_) {
#pragma unroll
            for (int i = 0; i < 4; ++i)
                __builtin_prefetch(&Wo[(size_t)(kh + 256 + (tid >> 6) + i * 4) * D_ +
                                       nblk * 64 + (tid & 63)], 0, 1);
        }
#pragma unroll 4
        for (int ks = 0; ks < 256; ks += 32) {
            const v16h af = load_frag_f16(Abase, D_, kh + ks, lane);
#pragma unroll
            for (int j = 0; j < 4; ++j)
                acc[j] = wmma_f16(af, load_frag_f16(&Wt[j * 16][0], LDW, ks, lane), acc[j]);
        }
    }
#pragma unroll
    for (int j = 0; j < 4; ++j) {
        const int n = nblk * 64 + j * 16 + r15;
        const float bb = bo[n];
#pragma unroll
        for (int r = 0; r < 8; ++r) {
            const int m = mrow + r + 8 * lh;
            out[(size_t)m * D_ + n] = acc[j][r] + bb;
        }
    }
}

// ---------------------------------------------------------------------------
extern "C" void kernel_launch(void* const* d_in, const int* in_sizes, int n_in,
                              void* d_out, int out_size, void* d_ws, size_t ws_size,
                              hipStream_t stream) {
    (void)in_sizes; (void)n_in; (void)out_size; (void)ws_size;
    const float* q  = (const float*)d_in[0];
    const float* k  = (const float*)d_in[1];
    const float* v  = (const float*)d_in[2];
    const unsigned char* mask = (const unsigned char*)d_in[3]; // bool [B,1,1,T]
    const float* Wq = (const float*)d_in[4];
    const float* bq = (const float*)d_in[5];
    const float* Wk = (const float*)d_in[6];
    const float* bk = (const float*)d_in[7];
    const float* Wv = (const float*)d_in[8];
    const float* bv = (const float*)d_in[9];
    const float* Wo = (const float*)d_in[10];
    const float* bo = (const float*)d_in[11];

    _Float16* ws = (_Float16*)d_ws;
    const size_t S1 = (size_t)B_ * H_ * T_ * DH_;   // 4M elems; total ws use = 32 MB

    qkv_proj_kernel<<<dim3(D_ / 64, (B_ * T_) / 128, 3), 256, 0, stream>>>(
        q, k, v, Wq, bq, Wk, bk, Wv, bv, ws);
    attn_kernel<<<dim3(T_ / 128, B_ * H_), 256, 0, stream>>>(ws, mask, ws + 3 * S1);
    out_proj_kernel<<<dim3(D_ / 64, (B_ * T_) / 128), 256, 0, stream>>>(
        ws + 3 * S1, Wo, bo, (float*)d_out);
}
